// MixtralSparseMoeBlock_53223234732070
// MI455X (gfx1250) — matrix operand order repaired and checked by reading, not compiled, not run on hardware
//
#include <hip/hip_runtime.h>
#include <math.h>

#define E_ 8
#define H_ 1024
#define F_ 3584
#define T_ 4096   // B*S = 2*2048

typedef __attribute__((ext_vector_type(16))) __bf16 bf16x16;
typedef __attribute__((ext_vector_type(8)))  float  fx8;

// ---------------- helpers ----------------

__device__ __forceinline__ __bf16 f2bf(float f) {
  unsigned u = __builtin_bit_cast(unsigned, f);
  u += 0x7fffu + ((u >> 16) & 1u);                 // round-to-nearest-even
  return __builtin_bit_cast(__bf16, (unsigned short)(u >> 16));
}

// A-operand (16x32 bf16) element offset inside a 512-element tile.
// ISA layout: lane<16 holds K {0..7,16..23}, lane>=16 holds K {8..15,24..31},
// row M = lane%16; VGPR v pair order -> elem index.
__device__ __forceinline__ int a_off(int row, int kk) {
  int hi   = ((kk & 15) >= 8) ? 16 : 0;
  int elem = ((kk >= 16) ? 8 : 0) + (kk & 7);
  return (row + hi) * 16 + elem;
}

// B-operand (32x16 bf16) element offset inside a 512-element tile.
// lanes 0-15 hold K {0..7,16..23} (col N = lane), lanes 16-31 hold K {8..15,24..31}.
__device__ __forceinline__ int b_off(int kk, int n) {
  int lane = (((kk & 15) >= 8) ? 16 : 0) + n;
  return lane * 16 + ((kk >= 16) ? 8 : 0) + (kk & 7);
}

// ---------------- kernels ----------------

__global__ __launch_bounds__(256)
void init_kernel(float* __restrict__ y, int* __restrict__ counts) {
  size_t i = (size_t)blockIdx.x * 256 + threadIdx.x;
  if (i < (size_t)T_ * H_) y[i] = 0.f;
  if (i < E_) counts[i] = 0;
}

// one wave per token: logits, softmax, top-2, normalized combine weights
__global__ __launch_bounds__(256)
void router_kernel(const float* __restrict__ x, const float* __restrict__ gw,
                   float* __restrict__ logits_out, float* __restrict__ combine) {
  const int lane = threadIdx.x & 31;
  const int t    = blockIdx.x * 8 + (threadIdx.x >> 5);
  float acc[E_] = {};
  const float* xr = x + (size_t)t * H_;
  for (int i = 0; i < H_ / 32; ++i) {
    int h = i * 32 + lane;
    float xv = xr[h];
    const float4* g4 = (const float4*)(gw + (size_t)h * E_);
    float4 ga = g4[0], gb = g4[1];
    acc[0] += xv * ga.x; acc[1] += xv * ga.y; acc[2] += xv * ga.z; acc[3] += xv * ga.w;
    acc[4] += xv * gb.x; acc[5] += xv * gb.y; acc[6] += xv * gb.z; acc[7] += xv * gb.w;
  }
  for (int off = 16; off >= 1; off >>= 1)
    for (int e = 0; e < E_; ++e) acc[e] += __shfl_xor(acc[e], off, 32);
  if (lane == 0) {
    float m = acc[0];
    for (int e = 1; e < E_; ++e) m = fmaxf(m, acc[e]);
    float p[E_], s = 0.f;
    for (int e = 0; e < E_; ++e) { p[e] = __expf(acc[e] - m); s += p[e]; }
    int a0 = 0;
    for (int e = 1; e < E_; ++e) if (p[e] > p[a0]) a0 = e;
    int a1 = -1;
    for (int e = 0; e < E_; ++e) { if (e == a0) continue; if (a1 < 0 || p[e] > p[a1]) a1 = e; }
    float pw0 = p[a0] / s, pw1 = p[a1] / s;
    float wsum = pw0 + pw1;
    for (int e = 0; e < E_; ++e) {
      float c = (e == a0) ? pw0 / wsum : (e == a1) ? pw1 / wsum : 0.f;
      combine[(size_t)t * E_ + e]    = c;
      logits_out[(size_t)t * E_ + e] = acc[e];
    }
  }
}

__global__ __launch_bounds__(256)
void build_lists(const float* __restrict__ combine, int* __restrict__ counts,
                 int* __restrict__ tokList, float* __restrict__ wtList) {
  int t = blockIdx.x * 256 + threadIdx.x;
  if (t >= T_) return;
  for (int e = 0; e < E_; ++e) {
    float c = combine[(size_t)t * E_ + e];
    if (c > 0.f) {
      int pos = atomicAdd(&counts[e], 1);
      tokList[(size_t)e * T_ + pos] = t;
      wtList [(size_t)e * T_ + pos] = c;
    }
  }
}

// f32 [E][K][N] -> bf16 in WMMA B-operand tile layout: [E][N/16][K/32][512]
__global__ __launch_bounds__(256)
void convert_swizzle(const float* __restrict__ src, __bf16* __restrict__ dst,
                     int K, int N) {
  long long i4 = (long long)blockIdx.x * 256 + threadIdx.x;
  long long total4 = (long long)E_ * K * N / 4;
  if (i4 >= total4) return;
  long long i = i4 * 4;
  long long kn = (long long)K * N;
  int e = (int)(i / kn);
  long long rem = i - (long long)e * kn;
  int k  = (int)(rem / N);
  int nn = (int)(rem - (long long)k * N);
  float4 v = *(const float4*)(src + i);
  float vv[4] = { v.x, v.y, v.z, v.w };
  int kt = k >> 5, kk = k & 31;
  for (int j = 0; j < 4; ++j) {
    int n = nn + j;
    int nt = n >> 4, nc = n & 15;
    long long off = (((long long)e * (N / 16) + nt) * (K / 32) + kt) * 512 + b_off(kk, nc);
    dst[off] = f2bf(vv[j]);
  }
}

// Fused per-expert SwiGLU MLP on a 32-token tile, entirely staged in LDS.
// LDS: xs 64KB (A-swizzled x rows) + inner 224KB (A-swizzled activations) + lists.
__global__ __launch_bounds__(256)
void moe_expert_kernel(const float* __restrict__ x,
                       const int*   __restrict__ tokList,
                       const float* __restrict__ wtList,
                       const int*   __restrict__ counts,
                       const __bf16* __restrict__ wup,
                       const __bf16* __restrict__ wgate,
                       const __bf16* __restrict__ wdown,
                       float* __restrict__ y) {
  extern __shared__ char smem[];
  __bf16* xs    = (__bf16*)smem;                       // 32 ktiles * 2 rt * 512
  __bf16* inner = (__bf16*)(smem + 65536);             // 112 ktiles * 2 rt * 512
  int*    toks  = (int*)  (smem + 65536 + 229376);
  float*  wts   = (float*)(smem + 65536 + 229376 + 128);

  const int e    = blockIdx.y;
  const int tile = blockIdx.x;
  const int n    = counts[e];
  if (tile * 32 >= n) return;

  const int tid  = threadIdx.x;
  const int lane = tid & 31;
  const int wave = tid >> 5;

  if (tid < 32) {
    int r  = tile * 32 + tid;
    int cl = (r < n) ? r : (n - 1);
    toks[tid] = tokList[(size_t)e * T_ + cl];
    wts[tid]  = (r < n) ? wtList[(size_t)e * T_ + cl] : 0.f;
  }
  __syncthreads();

  // gather 32 token rows, convert f32->bf16, store in A-operand swizzle
  for (int j = 0; j < 16; ++j) {
    int c   = tid + 256 * j;            // 4096 chunks of 8 elements
    int row = c >> 7;
    int kc  = (c & 127) * 8;
    const float4* p = (const float4*)(x + (size_t)toks[row] * H_ + kc);
    float4 f0 = p[0], f1 = p[1];
    union { __bf16 h[8]; uint4 u; } pk;
    pk.h[0] = f2bf(f0.x); pk.h[1] = f2bf(f0.y); pk.h[2] = f2bf(f0.z); pk.h[3] = f2bf(f0.w);
    pk.h[4] = f2bf(f1.x); pk.h[5] = f2bf(f1.y); pk.h[6] = f2bf(f1.z); pk.h[7] = f2bf(f1.w);
    int kt = kc >> 5, within = kc & 31, rt = row >> 4, rl = row & 15;
    int base = ((kt * 2 + rt) * 32) * 16 + a_off(rl, within);   // 8 contiguous elems
    *(uint4*)(xs + base) = pk.u;
  }
  __syncthreads();

  // ---- up/gate projections + SwiGLU -> inner (bf16, A-swizzled over K=F) ----
  for (int it = 0; it < 28; ++it) {                 // 28 * 8 waves * 16 cols = 3584
    int f0 = it * 128 + wave * 16;
    fx8 u0 = {}, u1 = {}, g0 = {}, g1 = {};
    const __bf16* bu = wup   + (((size_t)e * (F_ / 16) + (f0 >> 4)) * (H_ / 32)) * 512;
    const __bf16* bg = wgate + (((size_t)e * (F_ / 16) + (f0 >> 4)) * (H_ / 32)) * 512;
    for (int kt = 0; kt < H_ / 32; ++kt) {
      bf16x16 a0 = *(const bf16x16*)(xs + ((kt * 2 + 0) * 32 + lane) * 16);
      bf16x16 a1 = *(const bf16x16*)(xs + ((kt * 2 + 1) * 32 + lane) * 16);
      bf16x16 Bu = *(const bf16x16*)(bu + (size_t)kt * 512 + lane * 16);
      bf16x16 Bg = *(const bf16x16*)(bg + (size_t)kt * 512 + lane * 16);
      u0 = __builtin_amdgcn_wmma_f32_16x16x32_bf16(false, a0, false, Bu, (short)0, u0, false, false);
      u1 = __builtin_amdgcn_wmma_f32_16x16x32_bf16(false, a1, false, Bu, (short)0, u1, false, false);
      g0 = __builtin_amdgcn_wmma_f32_16x16x32_bf16(false, a0, false, Bg, (short)0, g0, false, false);
      g1 = __builtin_amdgcn_wmma_f32_16x16x32_bf16(false, a1, false, Bg, (short)0, g1, false, false);
    }
    int f  = f0 + (lane & 15);
    int kt2 = f >> 5, w2 = f & 31;
    int rowadd = (lane >= 16) ? 8 : 0;
    for (int r = 0; r < 8; ++r) {
      float s0 = u0[r] / (1.f + __expf(-u0[r]));
      float s1 = u1[r] / (1.f + __expf(-u1[r]));
      int rl = r + rowadd;                          // row-low within each rt tile
      inner[((kt2 * 2 + 0) * 32) * 16 + a_off(rl, w2)] = f2bf(s0 * g0[r]);
      inner[((kt2 * 2 + 1) * 32) * 16 + a_off(rl, w2)] = f2bf(s1 * g1[r]);
    }
  }
  __syncthreads();

  // ---- down projection + weighted scatter ----
  for (int it = 0; it < 8; ++it) {                  // 8 * 8 waves * 16 cols = 1024
    int h0 = it * 128 + wave * 16;
    fx8 c0 = {}, c1 = {};
    const __bf16* bd = wdown + (((size_t)e * (H_ / 16) + (h0 >> 4)) * (F_ / 32)) * 512;
    for (int kt = 0; kt < F_ / 32; ++kt) {
      bf16x16 a0 = *(const bf16x16*)(inner + ((kt * 2 + 0) * 32 + lane) * 16);
      bf16x16 a1 = *(const bf16x16*)(inner + ((kt * 2 + 1) * 32 + lane) * 16);
      bf16x16 Bd = *(const bf16x16*)(bd + (size_t)kt * 512 + lane * 16);
      c0 = __builtin_amdgcn_wmma_f32_16x16x32_bf16(false, a0, false, Bd, (short)0, c0, false, false);
      c1 = __builtin_amdgcn_wmma_f32_16x16x32_bf16(false, a1, false, Bd, (short)0, c1, false, false);
    }
    int h = h0 + (lane & 15);
    int rowadd = (lane >= 16) ? 8 : 0;
    for (int r = 0; r < 8; ++r) {
      int row0 = r + rowadd;
      int row1 = 16 + r + rowadd;
      if (tile * 32 + row0 < n)
        atomicAdd(&y[(size_t)toks[row0] * H_ + h], c0[r] * wts[row0]);
      if (tile * 32 + row1 < n)
        atomicAdd(&y[(size_t)toks[row1] * H_ + h], c1[r] * wts[row1]);
    }
  }
}

// ---------------- launcher ----------------

extern "C" void kernel_launch(void* const* d_in, const int* in_sizes, int n_in,
                              void* d_out, int out_size, void* d_ws, size_t ws_size,
                              hipStream_t stream) {
  (void)in_sizes; (void)n_in; (void)out_size; (void)ws_size;
  const float* x   = (const float*)d_in[0];   // [B,S,H]
  const float* gw  = (const float*)d_in[1];   // [H,E]
  const float* wup = (const float*)d_in[2];   // [E,H,F]
  const float* wgt = (const float*)d_in[3];   // [E,H,F]
  const float* wdn = (const float*)d_in[4];   // [E,F,H]

  float* y      = (float*)d_out;              // [T,H]
  float* logits = y + (size_t)T_ * H_;        // [T,E]

  char* ws = (char*)d_ws;
  float*  combine = (float*)ws;                        // T*E*4   = 131072
  int*    counts  = (int*)  (ws + 131072);             // padded to 512
  int*    tokList = (int*)  (ws + 131584);             // E*T*4   = 131072
  float*  wtList  = (float*)(ws + 262656);             // E*T*4   = 131072
  __bf16* wup_sw  = (__bf16*)(ws + 393728);            // E*H*F bf16
  __bf16* wgt_sw  = wup_sw + (size_t)E_ * H_ * F_;
  __bf16* wdn_sw  = wgt_sw + (size_t)E_ * H_ * F_;

  init_kernel<<<(T_ * H_ + 255) / 256, 256, 0, stream>>>(y, counts);
  router_kernel<<<T_ / 8, 256, 0, stream>>>(x, gw, logits, combine);
  build_lists<<<T_ / 256, 256, 0, stream>>>(combine, counts, tokList, wtList);

  long long tot4 = (long long)E_ * H_ * F_ / 4;
  int cblocks = (int)((tot4 + 255) / 256);
  convert_swizzle<<<cblocks, 256, 0, stream>>>(wup, wup_sw, H_, F_);
  convert_swizzle<<<cblocks, 256, 0, stream>>>(wgt, wgt_sw, H_, F_);
  convert_swizzle<<<cblocks, 256, 0, stream>>>(wdn, wdn_sw, F_, H_);

  dim3 grid(T_ / 32, E_);                     // 128 tiles x 8 experts, early-exit on count
  size_t smem = 65536 + 229376 + 128 + 128;   // 295,168 B of the WGP's 320 KB LDS
  moe_expert_kernel<<<grid, 256, smem, stream>>>(x, tokList, wtList, counts,
                                                 wup_sw, wgt_sw, wdn_sw, y);
}
